// FusedMultiHeadAttention_33844342293284
// MI455X (gfx1250) — matrix operand order repaired and checked by reading, not compiled, not run on hardware
//
#include <hip/hip_runtime.h>
#include <hip/hip_bf16.h>

// ---------------------------------------------------------------------------
// CDNA5 (gfx1250) fused GQA attention block, bf16 WMMA everywhere.
// wave32; WMMA shape 16x16x32 bf16 -> f32 accumulate.
// ---------------------------------------------------------------------------

typedef __attribute__((ext_vector_type(16))) __bf16 v16bf;
typedef __attribute__((ext_vector_type(8)))  __bf16 bfx8;
typedef __attribute__((ext_vector_type(8)))  float  v8f;

__device__ __forceinline__ __bf16 f2bf(float x) { return (__bf16)x; }

// ===========================================================================
// Kernel 1/4: bf16-WMMA GEMM  C(MxN) = A(MxK) @ B(KxN), fp32 in memory,
// converted to bf16 while staging. Compile-time M,N,K -> pure immediate
// address math. 256 threads = 8 waves; block tile 128x128; wave tile 32x64
// (2 M-subtiles x 4 N-subtiles = 8 WMMAs/chunk); K chunk 32.
// Ping-pong LDS double buffering: global loads of chunk i+1 are issued right
// after the barrier and stay in flight under the ds_load+WMMA of chunk i.
// ===========================================================================
#define GBM 128
#define GBN 128
#define GBK 32
#define ASTR 40   // As row stride in halfs (padded, 16B-aligned rows)
#define BSTR 40   // Bs row stride in halfs

template <int M, int N, int K>
__global__ __launch_bounds__(256) void gemm_bf16_wmma(
    const float* __restrict__ A, const float* __restrict__ B,
    float* __restrict__ C)
{
    __shared__ __bf16 As[2][GBM * ASTR];     // [m][k], row-major
    __shared__ __bf16 Bs[2][GBN * BSTR];     // [n][k], transposed

    const int t    = threadIdx.x;
    const int lane = t & 31, wid = t >> 5;
    const int wm   = wid & 3;                // 4 waves along M, 32 rows each
    const int wn   = wid >> 2;               // 2 waves along N, 64 cols each
    const int hi   = lane >> 4;              // lane half (wave32)
    const int lo   = lane & 15;
    const int m0   = blockIdx.y * GBM;
    const int n0   = blockIdx.x * GBN;

    // staging index maps (one thread stages 16 A floats + 16 B floats)
    const int am  = t >> 1;                  // 0..127 (A row)
    const int ak  = (t & 1) * 16;            // 0/16   (A k-start, 16 floats)
    const int bn  = t & 127;                 // 0..127 (B col)
    const int bk0 = (t >> 7) * 16;           // 0/16   (B k-start, 16 floats)

    const float* aptr = A + (size_t)(m0 + am) * K + ak;
    const float* bptr = B + (size_t)bk0 * N + n0 + bn;

    float4 ar0, ar1, ar2, ar3;
    float  br[16];

    auto gload = [&](int kb) {
        const float4* ap = reinterpret_cast<const float4*>(aptr + kb);
        ar0 = ap[0]; ar1 = ap[1]; ar2 = ap[2]; ar3 = ap[3];
        const float* bp = bptr + (size_t)kb * N;
#pragma unroll
        for (int j = 0; j < 16; j++) br[j] = bp[(size_t)j * N];
    };
    auto lstore = [&](int p) {
        bfx8 a0, a1;
        a0[0] = f2bf(ar0.x); a0[1] = f2bf(ar0.y); a0[2] = f2bf(ar0.z); a0[3] = f2bf(ar0.w);
        a0[4] = f2bf(ar1.x); a0[5] = f2bf(ar1.y); a0[6] = f2bf(ar1.z); a0[7] = f2bf(ar1.w);
        a1[0] = f2bf(ar2.x); a1[1] = f2bf(ar2.y); a1[2] = f2bf(ar2.z); a1[3] = f2bf(ar2.w);
        a1[4] = f2bf(ar3.x); a1[5] = f2bf(ar3.y); a1[6] = f2bf(ar3.z); a1[7] = f2bf(ar3.w);
        *reinterpret_cast<bfx8*>(&As[p][am * ASTR + ak])     = a0;
        *reinterpret_cast<bfx8*>(&As[p][am * ASTR + ak + 8]) = a1;
        bfx8 b0, b1;
#pragma unroll
        for (int j = 0; j < 8; j++) { b0[j] = f2bf(br[j]); b1[j] = f2bf(br[8 + j]); }
        *reinterpret_cast<bfx8*>(&Bs[p][bn * BSTR + bk0])     = b0;
        *reinterpret_cast<bfx8*>(&Bs[p][bn * BSTR + bk0 + 8]) = b1;
    };

    v8f acc[2][4];
#pragma unroll
    for (int u = 0; u < 2; u++)
#pragma unroll
        for (int st = 0; st < 4; st++)
#pragma unroll
            for (int i = 0; i < 8; i++) acc[u][st][i] = 0.0f;

    constexpr int NCH = K / GBK;
    gload(0);
    lstore(0);

    for (int it = 0; it < NCH; it++) {
        __syncthreads();                     // tile it visible; buffer it^1 free
        if (it + 1 < NCH) gload((it + 1) * GBK);   // in flight under compute
        if (it + 2 < NCH) {
            __builtin_prefetch(aptr + (it + 2) * GBK, 0, 3);
            __builtin_prefetch(bptr + (size_t)(it + 2) * GBK * N, 0, 3);
        }
        const int p = it & 1;

        // ---- A fragments: lane = row (lo), K = hi*8+i and 16+hi*8+i ----
        v16bf af[2];
#pragma unroll
        for (int u = 0; u < 2; u++) {
            const bfx8* pa = reinterpret_cast<const bfx8*>(
                &As[p][(wm * 32 + u * 16 + lo) * ASTR + hi * 8]);
            bfx8 x0 = pa[0], x1 = pa[2];     // +16 halfs
#pragma unroll
            for (int i = 0; i < 8; i++) { af[u][i] = x0[i]; af[u][8 + i] = x1[i]; }
        }
        // ---- 4 B fragments, 8 WMMAs ----
#pragma unroll
        for (int st = 0; st < 4; st++) {
            const bfx8* pb = reinterpret_cast<const bfx8*>(
                &Bs[p][(wn * 64 + st * 16 + lo) * BSTR + hi * 8]);
            bfx8 y0 = pb[0], y1 = pb[2];
            v16bf bf;
#pragma unroll
            for (int i = 0; i < 8; i++) { bf[i] = y0[i]; bf[8 + i] = y1[i]; }
            acc[0][st] = __builtin_amdgcn_wmma_f32_16x16x32_bf16(
                false, af[0], false, bf, (short)0, acc[0][st], false, false);
            acc[1][st] = __builtin_amdgcn_wmma_f32_16x16x32_bf16(
                false, af[1], false, bf, (short)0, acc[1][st], false, false);
        }
        if (it + 1 < NCH) lstore((it + 1) & 1);   // waits loadcnt implicitly
    }

    // ---- store C (f32). C-layout: vgpr i -> M = i + hi*8, N = lo ----
#pragma unroll
    for (int u = 0; u < 2; u++)
#pragma unroll
        for (int st = 0; st < 4; st++)
#pragma unroll
            for (int i = 0; i < 8; i++) {
                int mm = m0 + wm * 32 + u * 16 + i + hi * 8;
                int nn = n0 + wn * 64 + st * 16 + lo;
                C[(size_t)mm * N + nn] = acc[u][st][i];
            }
}

// ===========================================================================
// Kernel 2/4: RoPE + repack to bf16. qkv fp32 [b][s][4 kv][6 slot][64].
// Q (slots 0..3): rope, *0.125 (softmax scale folded in) -> Qb[b][h][s][64]
// K (slot 4):     rope                                   -> Kb[b][kv][s][64]
// V (slot 5):     pre-transposed                         -> Vt[b][kv][64][s]
// Grid (3, 4096): gridDim.y = (b,s) row, 768 threads cover 24 slots x 32
// pairs -- no integer division anywhere.
// ===========================================================================
__global__ __launch_bounds__(256) void rope_pack(
    const float* __restrict__ qkv, const int* __restrict__ curr_pos,
    __bf16* __restrict__ Qb, __bf16* __restrict__ Kb, __bf16* __restrict__ Vt)
{
    int v      = blockIdx.x * 256 + threadIdx.x;   // 0..767
    int j      = v & 31;                           // pair index 0..31
    int slot24 = v >> 5;                           // 0..23
    int bs     = blockIdx.y;                       // 0..4095
    int s      = bs & 2047;
    int b      = bs >> 11;
    int kvg    = slot24 >> 2;                      // wrong base? no: see below
    // slot24 = kvg*6 + slot, kvg in 0..3, slot in 0..5
    kvg        = (slot24 * 43) >> 8;               // fast /6 for 0..23
    int slot   = slot24 - kvg * 6;

    const float* row = qkv + (size_t)bs * 1536 + slot24 * 64;
    float x1 = row[2 * j], x2 = row[2 * j + 1];

    if (slot == 5) {  // V: transpose only
        size_t base = ((size_t)(b * 4 + kvg) * 64 + 2 * j) * 2048 + s;
        Vt[base]        = f2bf(x1);
        Vt[base + 2048] = f2bf(x2);
        return;
    }
    float pos      = (float)(s + curr_pos[0]);
    float inv_freq = __expf(-(float)(2 * j) * 0.14391156605213165f); // ln(1e4)/64
    float f = pos * inv_freq, sn, cs;
    __sincosf(f, &sn, &cs);
    float o1 = x1 * cs - x2 * sn;
    float o2 = x1 * sn + x2 * cs;

    if (slot < 4) {   // Q
        int h = kvg * 4 + slot;
        size_t base = ((size_t)(b * 16 + h) * 2048 + s) * 64 + 2 * j;
        Qb[base]     = f2bf(o1 * 0.125f);
        Qb[base + 1] = f2bf(o2 * 0.125f);
    } else {          // K
        size_t base = ((size_t)(b * 4 + kvg) * 2048 + s) * 64 + 2 * j;
        Kb[base]     = f2bf(o1);
        Kb[base + 1] = f2bf(o2);
    }
}

// ===========================================================================
// Kernel 3/4: causal flash attention, per-wave independent 16-row Q tile.
// Scores: 2 chained WMMAs over hd=64. V B-fragments are loaded from global
// *before* the softmax VALU block so the vmem latency hides under exp/shfl.
// P: C-layout -> A-layout through a private LDS tile guarded only by
// s_wait_dscnt (same-wave LDS is in-order; no workgroup barrier -- trip
// counts diverge per wave under causality). PV: 4 WMMAs per 32-key tile.
// ===========================================================================
#define PSTR 40
__global__ __launch_bounds__(256) void flash_attn(
    const __bf16* __restrict__ Qb, const __bf16* __restrict__ Kb,
    const __bf16* __restrict__ Vt, float* __restrict__ O)
{
    __shared__ __bf16 Pl[8 * 16 * PSTR];

    const int lane = threadIdx.x & 31, wid = threadIdx.x >> 5;
    const int hi = lane >> 4, lo = lane & 15;
    int gw = blockIdx.x * 8 + wid;            // 4096 wave tasks
    int qt = gw & 127, bh = gw >> 7;
    int h  = bh & 15,  b  = bh >> 4;
    int kv = h >> 2;
    int q0 = qt * 16;

    const __bf16* Qp = Qb + ((size_t)(b * 16 + h) * 2048) * 64;
    const __bf16* Kp = Kb + ((size_t)(b * 4 + kv) * 2048) * 64;
    const __bf16* Vp = Vt + ((size_t)(b * 4 + kv) * 64) * 2048;
    __bf16* P = Pl + wid * 16 * PSTR;

    // ---- Q A-fragments (hd=64 -> 2 chunks of K=32), loaded once ----
    v16bf qf[2];
#pragma unroll
    for (int c = 0; c < 2; c++) {
        const bfx8* p = reinterpret_cast<const bfx8*>(
            Qp + (size_t)(q0 + lo) * 64 + c * 32 + hi * 8);
        bfx8 a0 = p[0], a1 = p[2];
#pragma unroll
        for (int i = 0; i < 8; i++) { qf[c][i] = a0[i]; qf[c][8 + i] = a1[i]; }
    }

    float m[8], l[8];
    v8f acc[4];
#pragma unroll
    for (int i = 0; i < 8; i++) { m[i] = -1e30f; l[i] = 0.0f; }
#pragma unroll
    for (int tt = 0; tt < 4; tt++)
#pragma unroll
        for (int i = 0; i < 8; i++) acc[tt][i] = 0.0f;

    for (int kt = 0; kt < q0 + 16; kt += 32) {
        // ---- scores for two 16-key subtiles ----
        v8f sc[2];
#pragma unroll
        for (int sub = 0; sub < 2; sub++) {
            v16bf kf0, kf1;  // B-fragments: K^T chunks c=0,1 (contiguous in K[s][64])
            {
                const __bf16* kr = Kp + (size_t)(kt + sub * 16 + lo) * 64;
                const bfx8* p0 = reinterpret_cast<const bfx8*>(kr + hi * 8);
                const bfx8* p1 = reinterpret_cast<const bfx8*>(kr + 32 + hi * 8);
                bfx8 b0 = p0[0], b1 = p0[2], c0 = p1[0], c1 = p1[2];
#pragma unroll
                for (int i = 0; i < 8; i++) {
                    kf0[i] = b0[i]; kf0[8 + i] = b1[i];
                    kf1[i] = c0[i]; kf1[8 + i] = c1[i];
                }
            }
            v8f s;
#pragma unroll
            for (int i = 0; i < 8; i++) s[i] = 0.0f;
            s = __builtin_amdgcn_wmma_f32_16x16x32_bf16(false, qf[0], false, kf0,
                                                        (short)0, s, false, false);
            s = __builtin_amdgcn_wmma_f32_16x16x32_bf16(false, qf[1], false, kf1,
                                                        (short)0, s, false, false);
            // causal mask (softmax scale already folded into Q)
            int kcol = kt + sub * 16 + lo;
#pragma unroll
            for (int i = 0; i < 8; i++)
                if (kcol > q0 + i + hi * 8) s[i] = -1e30f;
            sc[sub] = s;
        }

        // ---- V B-fragments: issue global loads now, consume after softmax ----
        v16bf vf[4];
#pragma unroll
        for (int tt = 0; tt < 4; tt++) {
            const bfx8* p = reinterpret_cast<const bfx8*>(
                Vp + (size_t)(tt * 16 + lo) * 2048 + kt + hi * 8);
            bfx8 b0 = p[0], b1 = p[2];
#pragma unroll
            for (int i = 0; i < 8; i++) { vf[tt][i] = b0[i]; vf[tt][8 + i] = b1[i]; }
        }

        // ---- online softmax: stats per row via half-wave xor reductions ----
        float p0[8], p1[8];
#pragma unroll
        for (int i = 0; i < 8; i++) {
            float v = fmaxf(sc[0][i], sc[1][i]);
#pragma unroll
            for (int msk = 1; msk < 16; msk <<= 1)
                v = fmaxf(v, __shfl_xor(v, msk, 32));
            float mn = fmaxf(m[i], v);
            float r  = __expf(m[i] - mn);
            p0[i] = __expf(sc[0][i] - mn);
            p1[i] = __expf(sc[1][i] - mn);
            float rs = p0[i] + p1[i];
#pragma unroll
            for (int msk = 1; msk < 16; msk <<= 1)
                rs += __shfl_xor(rs, msk, 32);
            l[i] = l[i] * r + rs;
            m[i] = mn;
#pragma unroll
            for (int tt = 0; tt < 4; tt++) acc[tt][i] *= r;
        }

        // ---- P: C-layout -> LDS [q][k] -> A-layout (same wave only) ----
#pragma unroll
        for (int i = 0; i < 8; i++) {
            int rr = i + hi * 8;
            P[rr * PSTR + lo]      = f2bf(p0[i]);
            P[rr * PSTR + 16 + lo] = f2bf(p1[i]);
        }
        asm volatile("s_wait_dscnt 0" ::: "memory");
        v16bf pf;
        {
            const bfx8* p = reinterpret_cast<const bfx8*>(&P[lo * PSTR + hi * 8]);
            bfx8 a0 = p[0], a1 = p[2];
#pragma unroll
            for (int i = 0; i < 8; i++) { pf[i] = a0[i]; pf[8 + i] = a1[i]; }
        }

        // ---- PV: O[16x64] += P[16x32] @ V[32x64] (4 N-subtiles) ----
#pragma unroll
        for (int tt = 0; tt < 4; tt++)
            acc[tt] = __builtin_amdgcn_wmma_f32_16x16x32_bf16(
                false, pf, false, vf[tt], (short)0, acc[tt], false, false);
    }

    // ---- epilogue: normalize, write fp32 [b][s][h*64+d] ----
#pragma unroll
    for (int i = 0; i < 8; i++) {
        float inv = 1.0f / l[i];
        int qrow = q0 + i + hi * 8;
        size_t base = ((size_t)(b * 2048) + qrow) * 1024 + h * 64;
#pragma unroll
        for (int tt = 0; tt < 4; tt++)
            O[base + tt * 16 + lo] = acc[tt][i] * inv;
    }
}

// ===========================================================================
// Launch: qkv GEMM -> rope/pack -> flash attention -> out GEMM.
// Workspace (~36 MB): qkv f32 (25.2MB, later aliased by attn_out f32 16.8MB),
// Qb bf16 8.4MB, Kb bf16 2.1MB, Vt bf16 2.1MB.
// ===========================================================================
extern "C" void kernel_launch(void* const* d_in, const int* in_sizes, int n_in,
                              void* d_out, int out_size, void* d_ws, size_t ws_size,
                              hipStream_t stream)
{
    const float* x        = (const float*)d_in[0];
    const float* Wqkv     = (const float*)d_in[1];
    const float* Wo       = (const float*)d_in[2];
    const int*   curr_pos = (const int*)d_in[3];
    float*       out      = (float*)d_out;

    char* ws = (char*)d_ws;
    float*  qkv  = (float*)(ws);                 // 4096 x 1536 f32
    float*  attn = (float*)(ws);                 // 4096 x 1024 f32 (aliases dead qkv)
    __bf16* Qbuf = (__bf16*)(ws + 25165824);     // [2][16][2048][64]
    __bf16* Kbuf = (__bf16*)(ws + 33554432);     // [2][4][2048][64]
    __bf16* Vtb  = (__bf16*)(ws + 35651584);     // [2][4][64][2048]

    // 1) qkv = x @ Wqkv            (M=4096, N=1536, K=1024)
    gemm_bf16_wmma<4096, 1536, 1024>
        <<<dim3(1536 / GBN, 4096 / GBM), 256, 0, stream>>>(x, Wqkv, qkv);
    // 2) RoPE + bf16 repack (V transposed)
    rope_pack<<<dim3(3, 4096), 256, 0, stream>>>(qkv, curr_pos, Qbuf, Kbuf, Vtb);
    // 3) causal flash attention -> attn (aliases qkv, which is dead by now)
    flash_attn<<<512, 256, 0, stream>>>(Qbuf, Kbuf, Vtb, attn);
    // 4) out = attn @ Wo           (M=4096, N=1024, K=1024)
    gemm_bf16_wmma<4096, 1024, 1024>
        <<<dim3(1024 / GBN, 4096 / GBM), 256, 0, stream>>>(attn, Wo, out);
}
